// Mamba2dBlock_40527311405223
// MI455X (gfx1250) — matrix operand order repaired and checked by reading, not compiled, not run on hardware
//
#include <hip/hip_runtime.h>
#include <hip/hip_bf16.h>

// ---------------------------------------------------------------------------
// Mamba2D (VMamba cross-scan) block for MI455X / gfx1250.
// wave32 + WMMA bf16 + Tensor Data Mover (TDM) LDS staging.
// ---------------------------------------------------------------------------

typedef __attribute__((ext_vector_type(16))) __bf16 v16bf;
typedef __attribute__((ext_vector_type(8)))  __bf16 v8bf;
typedef __attribute__((ext_vector_type(8)))  float  v8f;
typedef __attribute__((ext_vector_type(4)))  unsigned int v4u;
typedef __attribute__((ext_vector_type(8)))  int v8i;
typedef __attribute__((ext_vector_type(4)))  int v4i;

#define BB      4
#define HW      64
#define LL      4096          // HW*HW
#define DMODEL  256
#define DSTATE  16
#define DINNER  512
#define NSEQ    16            // 4 directions * 4 batches
#define MROWS   (NSEQ * LL)   // 65536 token rows through the mamba core

// LDS layout: sA = 128 rows x 40 bf16, then sB = 64 rows x 40 bf16
#define LDS_ROW   40
#define SA_ELEMS  (128 * LDS_ROW)
#define SB_BYTE   (SA_ELEMS * 2)     // 10240
#define LDS_TOT   (SA_ELEMS + 64 * LDS_ROW)

// TDM available only in the gfx1250 device pass and when the builtin exists.
#if defined(__HIP_DEVICE_COMPILE__) && defined(__gfx1250__) && \
    __has_builtin(__builtin_amdgcn_tensor_load_to_lds)
#define HAS_TDM 1
#else
#define HAS_TDM 0
#endif

#if HAS_TDM
// 2D tile DMA: global (row-major, bf16, row stride ld_elems) -> LDS, with
// D# pad: after every 64 bytes stored insert 16 bytes (32+8 bf16 row stride).
// tensor dims are the remaining extents so edge tiles zero-fill (ISA 8.2 OOB).
__device__ __forceinline__ void tdm_load_2d(unsigned lds_byte_off,
                                            const void* gptr,
                                            unsigned rem_k, unsigned rem_rows,
                                            unsigned ld_elems,
                                            unsigned tile_k, unsigned tile_rows) {
  const unsigned long long ga = (unsigned long long)(uintptr_t)gptr;
  v4u g0;
  g0.x = 1u;                                   // count=1, is_restore=0
  g0.y = lds_byte_off;                         // lds_addr
  g0.z = (unsigned)ga;                         // global_addr[31:0]
  g0.w = ((unsigned)(ga >> 32) & 0x01FFFFFFu)  // global_addr[56:32]
         | (2u << 30);                         // type = 2 ("image")
  // group1: data_size=1 (2B), pad_enable, pad_interval=3 (64B), pad_amount=3 (16B)
  const unsigned d0 = (1u << 16) | (1u << 20) | (3u << 22) | (3u << 25);
  const unsigned long long s0 = (unsigned long long)ld_elems;  // dim0 stride
  v8i g1;
  g1[0] = (int)d0;
  g1[1] = (int)((rem_k & 0xFFFFu) << 16);                       // tensor_dim0[15:0]
  g1[2] = (int)((rem_k >> 16) | ((rem_rows & 0xFFFFu) << 16));  // dim0 hi | dim1 lo
  g1[3] = (int)((rem_rows >> 16) | (tile_k << 16));             // dim1 hi | tile_dim0
  g1[4] = (int)(tile_rows & 0xFFFFu);                           // tile_dim1, tile_dim2=0
  g1[5] = (int)(unsigned)(s0 & 0xFFFFFFFFu);                    // dim0_stride[31:0]
  g1[6] = (int)(unsigned)((s0 >> 32) & 0xFFFFu);                // dim0_stride[47:32]
  g1[7] = 0;
  v4i z4 = {0, 0, 0, 0};
#if __clang_major__ >= 23
  v8i z8 = {0, 0, 0, 0, 0, 0, 0, 0};
  __builtin_amdgcn_tensor_load_to_lds(g0, g1, z4, z4, z8, 0);
#else
  __builtin_amdgcn_tensor_load_to_lds(g0, g1, z4, z4, 0);
#endif
}
#endif  // HAS_TDM

// ---------------------------------------------------------------------------
// cross-scan gather: logical mamba row m -> source row of x (4,4096,256)
// ---------------------------------------------------------------------------
__device__ __forceinline__ int map_row_scan(int m) {
  int seq = m >> 12;
  int l   = m & 4095;
  int b   = seq & 3;
  int dir = seq >> 2;
  int lm;
  if (dir == 0)      lm = l;
  else if (dir == 1) { int r = l >> 6, c = l & 63; lm = ((63 - c) << 6) + r; }
  else if (dir == 2) lm = 4095 - l;
  else { int l2 = 4095 - l; int r = l2 >> 6, c = l2 & 63; lm = ((63 - c) << 6) + r; }
  return (b << 12) + lm;
}

// ---------------------------------------------------------------------------
// f32 -> bf16 convert (for weights)
// ---------------------------------------------------------------------------
__global__ __launch_bounds__(256) void cvt_f32_bf16(const float* __restrict__ s,
                                                    __bf16* __restrict__ d, int n) {
  int i = blockIdx.x * 256 + threadIdx.x;
  if (i < n) d[i] = (__bf16)s[i];
}

// ---------------------------------------------------------------------------
// Generic WMMA GEMM:  C(M,Nact) = A(M,Kact) * W(Nact,Kact)^T  (+ epilogue)
//   block tile 128x64, BK=32, 256 threads = 8 waves (4x2), wave tile 32x32.
//   AMAP: 0 = identity row map, 1 = cross-scan gather from x
//   AF32: A source is f32 (converted to bf16 while staging into LDS)
//   EPI : 0 = f32 | 1 = bf16 | 2 = softplus(acc+bias)->bf16 | 3 = acc+bias+res->f32
//   W tile always staged by TDM when available; A tile staged by TDM when it is
//   bf16 with identity row map.
// ---------------------------------------------------------------------------
template <int AMAP, bool AF32, int EPI>
__global__ __launch_bounds__(256) void gemm_wmma_kernel(
    const void* __restrict__ Av, const __bf16* __restrict__ W,
    void* __restrict__ Cv, const float* __restrict__ bias,
    const float* __restrict__ res,
    int M, int Kact, int lda, int Nact, int ldc) {
  __shared__ alignas(16) __bf16 smem[LDS_TOT];   // sA @ 0, sB @ SB_BYTE

  constexpr bool kTdmW = (HAS_TDM != 0);
  constexpr bool kTdmA = (HAS_TDM != 0) && (!AF32) && (AMAP == 0);

  const int tid  = threadIdx.x;
  const int lane = tid & 31;
  const int wave = tid >> 5;
  const int wr   = wave >> 1;
  const int wc   = wave & 1;
  const int half = lane >> 4;
  const int l16  = lane & 15;

  const int m0 = blockIdx.y * 128;
  const int n0 = blockIdx.x * 64;

  v8f c[2][2];
#pragma unroll
  for (int i = 0; i < 2; ++i)
#pragma unroll
    for (int j = 0; j < 2; ++j)
#pragma unroll
      for (int e = 0; e < 8; ++e) c[i][j][e] = 0.0f;

  const int numKt = (Kact + 31) >> 5;
  const int ar = tid >> 1, ach = (tid & 1) << 4;

  for (int kt = 0; kt < numKt; ++kt) {
    const int k0 = kt << 5;

#if HAS_TDM
    // ---- TDM staging: wave 0 issues descriptors, waits TENSORcnt ----
    if (wave == 0) {
      if (kTdmA)
        tdm_load_2d(0u, (const __bf16*)Av + (size_t)m0 * lda + k0,
                    (unsigned)(Kact - k0), (unsigned)(M - m0),
                    (unsigned)lda, 32u, 128u);
      if (kTdmW)
        tdm_load_2d(SB_BYTE, W + (size_t)n0 * Kact + k0,
                    (unsigned)(Kact - k0), (unsigned)(Nact - n0),
                    (unsigned)Kact, 32u, 64u);
      __builtin_amdgcn_s_wait_tensorcnt(0);
    }
#endif

    // ---- manual A staging (f32 source and/or cross-scan gather) ----
    if (!kTdmA) {
      const int gm = m0 + ar;
      const int gk = k0 + ach;
      __bf16 tmp[16];
      const bool ok = (gm < M) && (gk + 16 <= Kact);
      if (ok) {
        const int arow = (AMAP == 1) ? map_row_scan(gm) : gm;
        if (AF32) {
          const float* ap = (const float*)Av + (size_t)arow * lda + gk;
#pragma unroll
          for (int i = 0; i < 16; ++i) tmp[i] = (__bf16)ap[i];
          if (kt + 1 < numKt) __builtin_prefetch(ap + 32, 0, 0);
        } else {
          const __bf16* ap = (const __bf16*)Av + (size_t)arow * lda + gk;
#pragma unroll
          for (int i = 0; i < 16; ++i) tmp[i] = ap[i];
          if (kt + 1 < numKt) __builtin_prefetch(ap + 32, 0, 0);
        }
      } else {
#pragma unroll
        for (int i = 0; i < 16; ++i) tmp[i] = (__bf16)0.0f;
      }
#pragma unroll
      for (int i = 0; i < 16; ++i) smem[ar * LDS_ROW + ach + i] = tmp[i];
    }
    // ---- manual W staging (only without TDM) ----
    if (!kTdmW) {
      if (tid < 128) {
        const int br = tid >> 1, bch = (tid & 1) << 4;
        const int gn = n0 + br, gk = k0 + bch;
        __bf16 tmp[16];
        const bool ok = (gn < Nact) && (gk + 16 <= Kact);
        if (ok) {
          const __bf16* wp = W + (size_t)gn * Kact + gk;
#pragma unroll
          for (int i = 0; i < 16; ++i) tmp[i] = wp[i];
        } else {
#pragma unroll
          for (int i = 0; i < 16; ++i) tmp[i] = (__bf16)0.0f;
        }
#pragma unroll
        for (int i = 0; i < 16; ++i) smem[SA_ELEMS + br * LDS_ROW + bch + i] = tmp[i];
      }
    }
    __syncthreads();

    // ---- fragment loads (contiguous k = 16*half + e for A and B) ----
    v16bf a[2], b[2];
#pragma unroll
    for (int i = 0; i < 2; ++i) {
      const int rr = wr * 32 + i * 16 + l16;            // A row = lane%16
      const __bf16* base = smem + rr * LDS_ROW;
      v8bf p0 = *(const v8bf*)(base + half * 16);       // K 0..7 / 16..23
      v8bf p1 = *(const v8bf*)(base + half * 16 + 8);   // K 8..15 / 24..31
#pragma unroll
      for (int e = 0; e < 8; ++e) { a[i][e] = p0[e]; a[i][8 + e] = p1[e]; }
    }
#pragma unroll
    for (int j = 0; j < 2; ++j) {
      const int nn = wc * 32 + j * 16 + l16;            // B col = lane%16
      const __bf16* base = smem + SA_ELEMS + nn * LDS_ROW;
      v8bf q0 = *(const v8bf*)(base + half * 16);       // 16 contiguous K
      v8bf q1 = *(const v8bf*)(base + half * 16 + 8);
#pragma unroll
      for (int e = 0; e < 8; ++e) { b[j][e] = q0[e]; b[j][8 + e] = q1[e]; }
    }
#pragma unroll
    for (int i = 0; i < 2; ++i)
#pragma unroll
      for (int j = 0; j < 2; ++j)
        c[i][j] = __builtin_amdgcn_wmma_f32_16x16x32_bf16(
            false, a[i], false, b[j], (short)0, c[i][j], false, false);
    __syncthreads();
  }

  // ---- epilogue: C/D layout -> lane%16 = N, VGPR e + 8*half = M ----
#pragma unroll
  for (int i = 0; i < 2; ++i)
#pragma unroll
    for (int j = 0; j < 2; ++j) {
      const int gn = n0 + wc * 32 + j * 16 + l16;
      if (gn >= Nact) continue;
#pragma unroll
      for (int e = 0; e < 8; ++e) {
        const int gm = m0 + wr * 32 + i * 16 + half * 8 + e;
        if (gm >= M) continue;
        const size_t idx = (size_t)gm * ldc + gn;
        const float val = c[i][j][e];
        if (EPI == 0) {
          ((float*)Cv)[idx] = val;
        } else if (EPI == 1) {
          ((__bf16*)Cv)[idx] = (__bf16)val;
        } else if (EPI == 2) {
          const float t  = val + bias[gn];
          const float sp = (t > 20.0f) ? t : __logf(1.0f + __expf(t));
          ((__bf16*)Cv)[idx] = (__bf16)sp;
        } else {
          ((float*)Cv)[idx] = val + bias[gn] + res[idx];
        }
      }
    }
}

// ---------------------------------------------------------------------------
// depthwise causal conv1d (k=4) + bias + SiLU over xc half of xz
// ---------------------------------------------------------------------------
__global__ __launch_bounds__(256) void conv_silu_kernel(
    const __bf16* __restrict__ xz, const float* __restrict__ cw,
    const float* __restrict__ cb, __bf16* __restrict__ xcc) {
  const size_t idx = (size_t)blockIdx.x * 256 + threadIdx.x;  // 16*4096*512
  const int d = (int)(idx & 511);
  const size_t row = idx >> 9;  // seq*4096 + l
  const int l = (int)(row & 4095);
  float acc = cb[d];
#pragma unroll
  for (int k = 0; k < 4; ++k) {
    const int back = 3 - k;
    if (l - back >= 0)
      acc += cw[d * 4 + k] * (float)xz[(row - back) * 1024 + d];
  }
  const float s = acc / (1.0f + __expf(-acc));
  xcc[row * 512 + d] = (__bf16)s;
}

// ---------------------------------------------------------------------------
// selective scan: one channel per lane; h[16] and A-row[16] in VGPRs.
// fuses D-skip and SiLU(z) gating. grid (2,16), block 256.
// ---------------------------------------------------------------------------
__global__ __launch_bounds__(256) void scan_kernel(
    const __bf16* __restrict__ dtb, const __bf16* __restrict__ xcc,
    const __bf16* __restrict__ xz, const float* __restrict__ xdbl,
    const float* __restrict__ A_log, const float* __restrict__ D_skip,
    __bf16* __restrict__ yact) {
  const int d   = blockIdx.x * 256 + threadIdx.x;  // 0..511
  const int seq = blockIdx.y;                      // 0..15
  float Arow[16];
#pragma unroll
  for (int s = 0; s < 16; ++s) Arow[s] = -__expf(A_log[d * 16 + s]);
  const float Dd = D_skip[d];
  float h[16];
#pragma unroll
  for (int s = 0; s < 16; ++s) h[s] = 0.0f;

  const size_t rbase = (size_t)seq * LL;
  for (int l = 0; l < LL; ++l) {
    const size_t row = rbase + l;
    const float dtv = (float)dtb[row * 512 + d];
    const float xv  = (float)xcc[row * 512 + d];
    const float zv  = (float)xz[row * 1024 + 512 + d];
    const float4* Bp = (const float4*)(xdbl + row * 64 + 16);
    const float4* Cp = (const float4*)(xdbl + row * 64 + 32);
    float Bv[16], Cvv[16];
#pragma unroll
    for (int q = 0; q < 4; ++q) {
      const float4 bq = Bp[q], cq = Cp[q];
      Bv[4 * q + 0] = bq.x; Bv[4 * q + 1] = bq.y; Bv[4 * q + 2] = bq.z; Bv[4 * q + 3] = bq.w;
      Cvv[4 * q + 0] = cq.x; Cvv[4 * q + 1] = cq.y; Cvv[4 * q + 2] = cq.z; Cvv[4 * q + 3] = cq.w;
    }
    const float dtx = dtv * xv;
    float y = 0.0f;
#pragma unroll
    for (int s = 0; s < 16; ++s) {
      const float dA = __expf(dtv * Arow[s]);
      h[s] = dA * h[s] + dtx * Bv[s];
      y += h[s] * Cvv[s];
    }
    const float yy = y + Dd * xv;
    const float sz = zv / (1.0f + __expf(-zv));
    yact[row * 512 + d] = (__bf16)(yy * sz);
  }
}

// ---------------------------------------------------------------------------
// merge 4 directions (inverse flips/transposes) + LayerNorm over 256 channels
// ---------------------------------------------------------------------------
__global__ __launch_bounds__(256) void merge_ln_kernel(
    const float* __restrict__ Y, const float* __restrict__ ln_g,
    const float* __restrict__ ln_b, __bf16* __restrict__ out) {
  __shared__ float red[256];
  const int b = blockIdx.x >> 12;
  const int l = blockIdx.x & 4095;
  const int i = l >> 6, j = l & 63;
  const int d = threadIdx.x;
  const int l2 = (j << 6) + (63 - i);
  const int l3 = 4095 - l;
  const int l4 = 4095 - l2;
  float v = Y[((size_t)(0 * 4 + b) * LL + l)  * 256 + d]
          + Y[((size_t)(1 * 4 + b) * LL + l2) * 256 + d]
          + Y[((size_t)(2 * 4 + b) * LL + l3) * 256 + d]
          + Y[((size_t)(3 * 4 + b) * LL + l4) * 256 + d];
  red[d] = v;
  __syncthreads();
  for (int s = 128; s > 0; s >>= 1) {
    if (d < s) red[d] += red[d + s];
    __syncthreads();
  }
  const float mu = red[0] * (1.0f / 256.0f);
  __syncthreads();
  const float dv = v - mu;
  red[d] = dv * dv;
  __syncthreads();
  for (int s = 128; s > 0; s >>= 1) {
    if (d < s) red[d] += red[d + s];
    __syncthreads();
  }
  const float var = red[0] * (1.0f / 256.0f);
  const float nrm = dv * rsqrtf(var + 1e-5f) * ln_g[d] + ln_b[d];
  out[((size_t)b * LL + l) * 256 + d] = (__bf16)nrm;
}

// ---------------------------------------------------------------------------
// host launcher
// ---------------------------------------------------------------------------
extern "C" void kernel_launch(void* const* d_in, const int* in_sizes, int n_in,
                              void* d_out, int out_size, void* d_ws, size_t ws_size,
                              hipStream_t stream) {
  (void)in_sizes; (void)n_in; (void)out_size; (void)ws_size;
  const float* x         = (const float*)d_in[0];   // (4,4096,256)
  const float* in_proj_w = (const float*)d_in[1];   // (1024,256)
  const float* conv_w    = (const float*)d_in[2];   // (512,1,4)
  const float* conv_b    = (const float*)d_in[3];   // (512,)
  const float* x_proj_w  = (const float*)d_in[4];   // (48,512)
  const float* dt_proj_w = (const float*)d_in[5];   // (512,16)
  const float* dt_proj_b = (const float*)d_in[6];   // (512,)
  const float* A_log     = (const float*)d_in[7];   // (512,16)
  const float* D_skip    = (const float*)d_in[8];   // (512,)
  const float* mamba_o_w = (const float*)d_in[9];   // (256,512)
  const float* ln_g      = (const float*)d_in[10];  // (256,)
  const float* ln_b      = (const float*)d_in[11];  // (256,)
  const float* blk_out_w = (const float*)d_in[12];  // (256,256)
  const float* blk_out_b = (const float*)d_in[13];  // (256,)
  float* out = (float*)d_out;                       // (4,4096,256)

  char* ws = (char*)d_ws;
  size_t off = 0;
  auto alloc = [&](size_t bytes) -> void* {
    void* p = ws + off;
    off += (bytes + 255) & ~(size_t)255;
    return p;
  };
  __bf16* wb_in = (__bf16*)alloc((size_t)1024 * 256 * 2);
  __bf16* wb_xp = (__bf16*)alloc((size_t)48 * 512 * 2);
  __bf16* wb_dt = (__bf16*)alloc((size_t)512 * 16 * 2);
  __bf16* wb_mo = (__bf16*)alloc((size_t)256 * 512 * 2);
  __bf16* wb_bo = (__bf16*)alloc((size_t)256 * 256 * 2);
  __bf16* xz    = (__bf16*)alloc((size_t)MROWS * 1024 * 2);  // in_proj out
  __bf16* xcc   = (__bf16*)alloc((size_t)MROWS * 512 * 2);   // conv+silu out
  float*  xdbl  = (float*) alloc((size_t)MROWS * 64 * 4);    // x_proj out (pad 48->64)
  __bf16* dtb   = (__bf16*)alloc((size_t)MROWS * 512 * 2);   // softplus(dt) out
  __bf16* yact  = (__bf16*)alloc((size_t)MROWS * 512 * 2);   // scan out (gated)
  float*  ym    = (float*) alloc((size_t)MROWS * 256 * 4);   // mamba_out out
  __bf16* nrm   = (__bf16*)alloc((size_t)BB * LL * 256 * 2); // merged+LN

  auto cvt = [&](const float* s, __bf16* d, int n) {
    cvt_f32_bf16<<<(n + 255) / 256, 256, 0, stream>>>(s, d, n);
  };
  cvt(in_proj_w, wb_in, 1024 * 256);
  cvt(x_proj_w,  wb_xp, 48 * 512);
  cvt(dt_proj_w, wb_dt, 512 * 16);
  cvt(mamba_o_w, wb_mo, 256 * 512);
  cvt(blk_out_w, wb_bo, 256 * 256);

  // 1) in_proj with fused cross-scan gather: xz = seqs @ in_proj_w^T
  gemm_wmma_kernel<1, true, 1><<<dim3(16, 512), 256, 0, stream>>>(
      x, wb_in, xz, nullptr, nullptr, MROWS, 256, 256, 1024, 1024);

  // 2) depthwise causal conv + SiLU
  conv_silu_kernel<<<(MROWS * 512) / 256, 256, 0, stream>>>(xz, conv_w, conv_b, xcc);

  // 3) x_proj: x_dbl = xcc @ x_proj_w^T  (N=48, padded stride 64; A via TDM)
  gemm_wmma_kernel<0, false, 0><<<dim3(1, 512), 256, 0, stream>>>(
      xcc, wb_xp, xdbl, nullptr, nullptr, MROWS, 512, 512, 48, 64);

  // 4) dt = softplus(dt_low @ dt_proj_w^T + dt_proj_b)  (K=16)
  gemm_wmma_kernel<0, true, 2><<<dim3(8, 512), 256, 0, stream>>>(
      xdbl, wb_dt, dtb, dt_proj_b, nullptr, MROWS, 16, 64, 512, 512);

  // 5) selective scan + D-skip + SiLU(z) gate
  scan_kernel<<<dim3(2, NSEQ), 256, 0, stream>>>(dtb, xcc, xz, xdbl, A_log, D_skip, yact);

  // 6) mamba out-proj: ym = yact @ mamba_out_w^T  (A via TDM)
  gemm_wmma_kernel<0, false, 0><<<dim3(4, 512), 256, 0, stream>>>(
      yact, wb_mo, ym, nullptr, nullptr, MROWS, 512, 512, 256, 256);

  // 7) merge 4 directions + LayerNorm
  merge_ln_kernel<<<BB * LL, 256, 0, stream>>>(ym, ln_g, ln_b, nrm);

  // 8) block out-proj + bias + residual: out = x + nrm @ blk_out_w^T + b
  gemm_wmma_kernel<0, false, 3><<<dim3(4, 128), 256, 0, stream>>>(
      nrm, wb_bo, out, blk_out_b, x, BB * LL, 256, 256, 256, 256);
}
